// Qwen3VLWithExpertModel_47132971107065
// MI455X (gfx1250) — matrix operand order repaired and checked by reading, not compiled, not run on hardware
//
#include <hip/hip_runtime.h>
#include <hip/hip_bf16.h>

typedef __bf16 bf16;
typedef __attribute__((ext_vector_type(16))) __bf16 v16bf;
typedef __attribute__((ext_vector_type(8)))  __bf16 v8bf;
typedef __attribute__((ext_vector_type(8)))  float  v8f;

#define B_    2
#define S0_   1024
#define S_    1344
#define H_    16
#define KV_   8
#define D_    128
#define HD_   2048   // H*D
#define KVD_  1024   // KV*D
#define EPS_  1e-6f
#define SCALE_ 0.08838834764831845f  // 1/sqrt(128)

// ---------------- device helpers ----------------

__device__ __forceinline__ bf16 f2bf(float f) {
  unsigned u = __builtin_bit_cast(unsigned, f);
  u += 0x7fffu + ((u >> 16) & 1u);            // round-to-nearest-even
  unsigned short h = (unsigned short)(u >> 16);
  return __builtin_bit_cast(bf16, h);
}

__device__ __forceinline__ v16bf pack16(v8bf lo, v8bf hh) {
  v16bf r;
#pragma unroll
  for (int i = 0; i < 8; ++i) { r[i] = lo[i]; r[8 + i] = hh[i]; }
  return r;
}

__device__ __forceinline__ v8f wmma_bf16(v16bf a, v16bf b, v8f c) {
  return __builtin_amdgcn_wmma_f32_16x16x32_bf16(false, a, false, b, (short)0, c,
                                                 false, false);
}

// ---------------- utility kernels ----------------

__global__ void k_f32_to_bf16(const float* __restrict__ in, bf16* __restrict__ out, long n) {
  long i = (long)blockIdx.x * blockDim.x + threadIdx.x;
  long stride = (long)gridDim.x * blockDim.x;
  for (; i < n; i += stride) out[i] = f2bf(in[i]);
}

__global__ __launch_bounds__(256) void k_rmsnorm_bf16(
    const float* __restrict__ x, const float* __restrict__ g,
    bf16* __restrict__ out, int hid) {
  __shared__ float red[256];
  const long row = blockIdx.x;
  const float* xr = x + row * (long)hid;
  float s = 0.f;
  for (int i = threadIdx.x; i < hid; i += 256) { float v = xr[i]; s += v * v; }
  red[threadIdx.x] = s;
  __syncthreads();
  for (int st = 128; st > 0; st >>= 1) {
    if ((int)threadIdx.x < st) red[threadIdx.x] += red[threadIdx.x + st];
    __syncthreads();
  }
  float sc = rsqrtf(red[0] / (float)hid + EPS_);
  bf16* orow = out + row * (long)hid;
  for (int i = threadIdx.x; i < hid; i += 256) orow[i] = f2bf(xr[i] * sc * g[i]);
}

// q/k head-RMSNorm + RoPE + scatter into packed attention buffers.
// grid = (M, H + 2*KV), block = 128 (=D). v is stored transposed [b,kv,D,S].
__global__ __launch_bounds__(128) void k_qkv_post(
    const float* __restrict__ qf, const float* __restrict__ kf,
    const float* __restrict__ vf, const float* __restrict__ qn,
    const float* __restrict__ kn, const int* __restrict__ pos_ids,
    bf16* __restrict__ qg, bf16* __restrict__ kg, bf16* __restrict__ vt,
    int Si, int start) {
  __shared__ float sm[128];
  const int m = blockIdx.x, y = blockIdx.y, d = threadIdx.x;
  const int b = m / Si, s = m - b * Si, sg = start + s;
  if (y >= H_ + KV_) {  // v path: just convert + transpose
    int kvh = y - H_ - KV_;
    float v = vf[(long)m * KVD_ + kvh * D_ + d];
    vt[(((long)b * KV_ + kvh) * D_ + d) * (long)S_ + sg] = f2bf(v);
    return;
  }
  float val; const float* gam; long obase; bf16* dst;
  if (y < H_) {
    val = qf[(long)m * HD_ + y * D_ + d]; gam = qn; dst = qg;
    obase = (((long)b * H_ + y) * S_ + sg) * (long)D_;
  } else {
    int kvh = y - H_;
    val = kf[(long)m * KVD_ + kvh * D_ + d]; gam = kn; dst = kg;
    obase = (((long)b * KV_ + kvh) * S_ + sg) * (long)D_;
  }
  sm[d] = val * val;
  __syncthreads();
  for (int st = 64; st > 0; st >>= 1) {
    if (d < st) sm[d] += sm[d + st];
    __syncthreads();
  }
  float ms = sm[0] / (float)D_;
  __syncthreads();
  float nv = val * rsqrtf(ms + EPS_) * gam[d];
  sm[d] = nv;
  __syncthreads();
  int dd = d & 63;
  float ang = (float)pos_ids[(long)b * S_ + sg] *
              __expf(-(float)(2 * dd) * (13.815510557964274f / 128.0f));
  float rot = (d < 64) ? -sm[d + 64] : sm[d - 64];
  float ov = nv * cosf(ang) + rot * sinf(ang);
  dst[obase + d] = f2bf(ov);
}

// ---------------- WMMA GEMM: C[M,N] = A[M,K](bf16) x W[N,K]^T(bf16) ----------------
// Block = 8 waves, tile 128 rows x 64 cols. The 64x32 W tile (shared by all 8
// waves) is staged through LDS, double-buffered: 256 threads x 16B = 4KB/chunk.
// Rows padded to 40 bf16 (80B) so the 16B fragment reads are bank-conflict-free.
// EPI: 0 = f32 store, 1 = f32 + residual, 2 = bf16 store, 3 = bf16 silu(aux)*acc
// A rows remapped: arow = (m/rpb)*bstr + roff + m%rpb  (for slicing attention rows)
template <int EPI>
__global__ __launch_bounds__(256) void k_gemm_wmma(
    const bf16* __restrict__ A, const bf16* __restrict__ W,
    float* __restrict__ Cf, bf16* __restrict__ Cb,
    const float* __restrict__ res, const float* __restrict__ aux,
    int N, int K, int rpb, int bstr, int roff) {
  __shared__ __align__(16) bf16 Wl[2][64][40];
  const int tid = threadIdx.x;
  const int lane = tid & 31;
  const int r = lane & 15, hi = lane >> 4;
  const int row0 = blockIdx.y * 128 + (tid >> 5) * 16;
  const int n0 = blockIdx.x * 64;
  const int mr = row0 + r;
  const long arow = (long)(mr / rpb) * bstr + roff + (mr % rpb);
  const bf16* ap = A + arow * (long)K;
  // cooperative W-tile staging: thread tid handles row tid/4, 16B chunk tid%4
  const int wrow = tid >> 2, wcol = (tid & 3) * 8;
  const bf16* wsrc = W + (long)(n0 + wrow) * K + wcol;
  v8f zero = {0.f, 0.f, 0.f, 0.f, 0.f, 0.f, 0.f, 0.f};
  v8f acc[4] = {zero, zero, zero, zero};

  *(v8bf*)&Wl[0][wrow][wcol] = *(const v8bf*)(wsrc);
  __syncthreads();
  int buf = 0;
  for (int k0 = 0; k0 < K; k0 += 32) {
    if (k0 + 32 < K)  // uniform: prefetch next W chunk into the other buffer
      *(v8bf*)&Wl[buf ^ 1][wrow][wcol] = *(const v8bf*)(wsrc + k0 + 32);
    v8bf alo = *(const v8bf*)(ap + k0 + hi * 8);
    v8bf ah2 = *(const v8bf*)(ap + k0 + 16 + hi * 8);
    v16bf af = pack16(alo, ah2);
#pragma unroll
    for (int t = 0; t < 4; ++t) {
      v8bf b0 = *(const v8bf*)&Wl[buf][t * 16 + r][hi * 8];
      v8bf b1 = *(const v8bf*)&Wl[buf][t * 16 + r][16 + hi * 8];
      acc[t] = wmma_bf16(af, pack16(b0, b1), acc[t]);
    }
    __syncthreads();
    buf ^= 1;
  }
#pragma unroll
  for (int t = 0; t < 4; ++t) {
    const int col = n0 + t * 16 + r;
#pragma unroll
    for (int j = 0; j < 8; ++j) {
      const long mrow = row0 + j + 8 * hi;
      const long idx = mrow * (long)N + col;
      float v = acc[t][j];
      if (EPI == 0) Cf[idx] = v;
      else if (EPI == 1) Cf[idx] = v + res[idx];
      else if (EPI == 2) Cb[idx] = f2bf(v);
      else { float gv = aux[idx]; Cb[idx] = f2bf((gv / (1.f + __expf(-gv))) * v); }
    }
  }
}

// ---------------- Flash attention (bf16 WMMA, online softmax) ----------------
// grid = (S/16/4, H, B), block = 128 (4 waves, one 16-query tile each)
__global__ __launch_bounds__(128) void k_flash_attn(
    const bf16* __restrict__ qg, const bf16* __restrict__ kg,
    const bf16* __restrict__ vt, const float* __restrict__ mask,
    bf16* __restrict__ att) {
  __shared__ __align__(32) bf16 Plds[4][16][32];
  const int wave = threadIdx.x >> 5, lane = threadIdx.x & 31;
  const int r = lane & 15, hi = lane >> 4;
  const int b = blockIdx.z, h = blockIdx.y;
  const int q0 = (blockIdx.x * 4 + wave) * 16;
  const int kv = h >> 1;  // GQA repeat = H/KV = 2
  const bf16* qb = qg + (((long)b * H_ + h) * S_ + q0 + r) * (long)D_;
  v16bf qfrag[4];
#pragma unroll
  for (int c = 0; c < 4; ++c) {
    v8bf lo = *(const v8bf*)(qb + c * 32 + hi * 16);
    v8bf hh = *(const v8bf*)(qb + c * 32 + hi * 16 + 8);
    qfrag[c] = pack16(lo, hh);
  }
  const bf16* kbase = kg + ((long)b * KV_ + kv) * (long)S_ * D_;
  const bf16* vbase = vt + ((long)b * KV_ + kv) * (long)D_ * S_;
  const float* mbase = mask + (long)b * S_ * S_;
  v8f zero = {0.f, 0.f, 0.f, 0.f, 0.f, 0.f, 0.f, 0.f};
  v8f o[8];
  float mrow[8], lrow[8];
#pragma unroll
  for (int i = 0; i < 8; ++i) { o[i] = zero; mrow[i] = -1e30f; lrow[i] = 0.f; }

  for (int kb0 = 0; kb0 < S_; kb0 += 32) {
    v8f sacc[2];
#pragma unroll
    for (int t = 0; t < 2; ++t) {
      v8f s = zero;
      const int key = kb0 + t * 16 + r;
      const bf16* kp = kbase + (long)key * D_;
#pragma unroll
      for (int c = 0; c < 4; ++c) {
        v16bf bfr = *(const v16bf*)(kp + c * 32 + hi * 16);
        s = wmma_bf16(qfrag[c], bfr, s);
      }
      const float* mp = mbase + key;
#pragma unroll
      for (int j = 0; j < 8; ++j)
        s[j] = s[j] * SCALE_ + mp[(long)(q0 + j + 8 * hi) * S_];
      sacc[t] = s;
    }
    // row max over 32 key-columns (shfl within 16-lane halves)
    float tm[8];
#pragma unroll
    for (int j = 0; j < 8; ++j) tm[j] = fmaxf(sacc[0][j], sacc[1][j]);
#pragma unroll
    for (int off = 1; off < 16; off <<= 1)
#pragma unroll
      for (int j = 0; j < 8; ++j) tm[j] = fmaxf(tm[j], __shfl_xor(tm[j], off, 32));
    float rs[8];
#pragma unroll
    for (int j = 0; j < 8; ++j) {
      float mn = fmaxf(mrow[j], tm[j]);
      float corr = __expf(mrow[j] - mn);
      float p0 = __expf(sacc[0][j] - mn);
      float p1 = __expf(sacc[1][j] - mn);
      mrow[j] = mn;
      rs[j] = p0 + p1;
      int q = j + 8 * hi;
      Plds[wave][q][r] = f2bf(p0);
      Plds[wave][q][16 + r] = f2bf(p1);
      lrow[j] *= corr;
#pragma unroll
      for (int dd = 0; dd < 8; ++dd) o[dd][j] *= corr;
    }
#pragma unroll
    for (int off = 1; off < 16; off <<= 1)
#pragma unroll
      for (int j = 0; j < 8; ++j) rs[j] += __shfl_xor(rs[j], off, 32);
#pragma unroll
    for (int j = 0; j < 8; ++j) lrow[j] += rs[j];
    // P tile back in A-fragment layout (same-wave LDS ops are in-order)
    asm volatile("s_wait_dscnt 0" ::: "memory");
    v8bf plo = *(const v8bf*)&Plds[wave][r][hi * 16];
    v8bf phi = *(const v8bf*)&Plds[wave][r][hi * 16 + 8];
    v16bf pf = pack16(plo, phi);
#pragma unroll
    for (int d0 = 0; d0 < 8; ++d0) {
      v16bf vfr = *(const v16bf*)(vbase + (long)(d0 * 16 + r) * S_ + kb0 + hi * 16);
      o[d0] = wmma_bf16(pf, vfr, o[d0]);
    }
  }
  bf16* ab = att + ((long)b * S_ + q0) * (long)HD_ + (long)h * D_;
#pragma unroll
  for (int d0 = 0; d0 < 8; ++d0)
#pragma unroll
    for (int j = 0; j < 8; ++j) {
      int q = j + 8 * hi;
      ab[(long)q * HD_ + d0 * 16 + r] = f2bf(o[d0][j] / lrow[j]);
    }
}

// ---------------- host orchestration ----------------

extern "C" void kernel_launch(void* const* d_in, const int* in_sizes, int n_in,
                              void* d_out, int out_size, void* d_ws, size_t ws_size,
                              hipStream_t stream) {
  (void)in_sizes; (void)n_in; (void)out_size; (void)ws_size;
  const float* xs[3] = {(const float*)d_in[0], (const float*)d_in[1], (const float*)d_in[2]};
  const float* mask = (const float*)d_in[3];
  const int* pos = (const int*)d_in[4];
  auto P = [&](int e, int j) { return (const float*)d_in[5 + e * 11 + j]; };
  // j: 0 ln1, 1 ln2, 2 wq, 3 wk, 4 wv, 5 wo, 6 qn, 7 kn, 8 wg, 9 wu, 10 wd
  const int hid[3] = {2048, 1024, 1024};
  const int ffn[3] = {8192, 4096, 4096};
  const int Si[3] = {1024, 256, 64};
  const int start[3] = {0, 1024, 1280};
  const int Mr[3] = {2048, 512, 128};

  char* p = (char*)d_ws;
  auto carve = [&](size_t bytes) -> void* {
    void* r = (void*)p;
    p += (bytes + 255) & ~(size_t)255;
    return r;
  };
  bf16 *wqb[3], *wkb[3], *wvb[3], *wob[3], *wgb[3], *wub[3], *wdb[3];
  for (int e = 0; e < 3; ++e) {
    wqb[e] = (bf16*)carve((size_t)HD_ * hid[e] * 2);
    wkb[e] = (bf16*)carve((size_t)KVD_ * hid[e] * 2);
    wvb[e] = (bf16*)carve((size_t)KVD_ * hid[e] * 2);
    wob[e] = (bf16*)carve((size_t)hid[e] * HD_ * 2);
    wgb[e] = (bf16*)carve((size_t)ffn[e] * hid[e] * 2);
    wub[e] = (bf16*)carve((size_t)ffn[e] * hid[e] * 2);
    wdb[e] = (bf16*)carve((size_t)hid[e] * ffn[e] * 2);
  }
  bf16* hA   = (bf16*)carve((size_t)2048 * 2048 * 2);
  float* qf  = (float*)carve((size_t)2048 * 2048 * 4);
  float* kf  = (float*)carve((size_t)2048 * 1024 * 4);
  float* vf  = (float*)carve((size_t)2048 * 1024 * 4);
  bf16* qall = (bf16*)carve((size_t)B_ * H_ * S_ * D_ * 2);
  bf16* kall = (bf16*)carve((size_t)B_ * KV_ * S_ * D_ * 2);
  bf16* vtr  = (bf16*)carve((size_t)B_ * KV_ * D_ * S_ * 2);
  bf16* attb = (bf16*)carve((size_t)B_ * S_ * HD_ * 2);
  float* obuf = (float*)carve((size_t)2048 * 2048 * 4);
  bf16* h2   = (bf16*)carve((size_t)2048 * 2048 * 2);
  float* gbuf = (float*)carve((size_t)2048 * 8192 * 4);
  bf16* mbuf = (bf16*)carve((size_t)2048 * 8192 * 2);

  auto conv = [&](const float* src, bf16* dst, long n) {
    int blocks = (int)((n + 2047) / 2048);
    if (blocks > 4096) blocks = 4096;
    k_f32_to_bf16<<<dim3(blocks), dim3(256), 0, stream>>>(src, dst, n);
  };
  for (int e = 0; e < 3; ++e) {
    conv(P(e, 2), wqb[e], (long)HD_ * hid[e]);
    conv(P(e, 3), wkb[e], (long)KVD_ * hid[e]);
    conv(P(e, 4), wvb[e], (long)KVD_ * hid[e]);
    conv(P(e, 5), wob[e], (long)hid[e] * HD_);
    conv(P(e, 8), wgb[e], (long)ffn[e] * hid[e]);
    conv(P(e, 9), wub[e], (long)ffn[e] * hid[e]);
    conv(P(e, 10), wdb[e], (long)hid[e] * ffn[e]);
  }
  auto gemm = [&](int epi, const bf16* A, const bf16* W, float* Cf, bf16* Cb,
                  const float* res, const float* aux, int M, int N, int K,
                  int rpb, int bstr, int roff) {
    dim3 grid(N / 64, M / 128), blk(256);
    switch (epi) {
      case 0: k_gemm_wmma<0><<<grid, blk, 0, stream>>>(A, W, Cf, Cb, res, aux, N, K, rpb, bstr, roff); break;
      case 1: k_gemm_wmma<1><<<grid, blk, 0, stream>>>(A, W, Cf, Cb, res, aux, N, K, rpb, bstr, roff); break;
      case 2: k_gemm_wmma<2><<<grid, blk, 0, stream>>>(A, W, Cf, Cb, res, aux, N, K, rpb, bstr, roff); break;
      default: k_gemm_wmma<3><<<grid, blk, 0, stream>>>(A, W, Cf, Cb, res, aux, N, K, rpb, bstr, roff); break;
    }
  };

  // ---- pre-attention: RMSNorm -> QKV projections -> qk-norm + RoPE ----
  for (int e = 0; e < 3; ++e) {
    const int M = Mr[e];
    k_rmsnorm_bf16<<<dim3(M), dim3(256), 0, stream>>>(xs[e], P(e, 0), hA, hid[e]);
    gemm(0, hA, wqb[e], qf, nullptr, nullptr, nullptr, M, HD_, hid[e], M, M, 0);
    gemm(0, hA, wkb[e], kf, nullptr, nullptr, nullptr, M, KVD_, hid[e], M, M, 0);
    gemm(0, hA, wvb[e], vf, nullptr, nullptr, nullptr, M, KVD_, hid[e], M, M, 0);
    k_qkv_post<<<dim3(M, H_ + 2 * KV_), dim3(128), 0, stream>>>(
        qf, kf, vf, P(e, 6), P(e, 7), pos, qall, kall, vtr, Si[e], start[e]);
  }
  // ---- joint flash attention over S=1344 ----
  k_flash_attn<<<dim3(S_ / 64, H_, B_), dim3(128), 0, stream>>>(qall, kall, vtr, mask, attb);

  // ---- post-attention: wo(+x) -> RMSNorm -> SwiGLU MLP -> wd(+o) -> d_out ----
  float* outp = (float*)d_out;
  const long ooff[3] = {0, (long)2 * 1024 * 2048, (long)2 * 1024 * 2048 + (long)2 * 256 * 1024};
  for (int e = 0; e < 3; ++e) {
    const int M = Mr[e];
    gemm(1, attb, wob[e], obuf, nullptr, xs[e], nullptr, M, hid[e], HD_, Si[e], S_, start[e]);
    k_rmsnorm_bf16<<<dim3(M), dim3(256), 0, stream>>>(obuf, P(e, 1), h2, hid[e]);
    gemm(0, h2, wgb[e], gbuf, nullptr, nullptr, nullptr, M, ffn[e], hid[e], M, M, 0);
    gemm(3, h2, wub[e], nullptr, mbuf, nullptr, gbuf, M, ffn[e], hid[e], M, M, 0);
    gemm(1, mbuf, wdb[e], outp + ooff[e], nullptr, obuf, nullptr, M, hid[e], ffn[e], M, M, 0);
  }
}
